// MogLSTM_14972255994023
// MI455X (gfx1250) — matrix operand order repaired and checked by reading, not compiled, not run on hardware
//
#include <hip/hip_runtime.h>

// MogLSTM: B=64, S=1024, I=H=512, gates=4H=2048, MOG_ITERS=5
#define Bn 64
#define Sn 1024
#define In 512
#define Hn 512
#define Gn 2048
#define LDSS 520              // LDS row stride in bf16 elems (pad -> conflict-free ds_load_b128)
#define NQ (512 * 512)        // elems in Q (and R)
#define NW (512 * 2048)       // elems in Wih (and Whh)

typedef unsigned short u16;
typedef __attribute__((ext_vector_type(16))) __bf16 v16bf;
typedef __attribute__((ext_vector_type(8)))  __bf16 v8bf;
typedef __attribute__((ext_vector_type(8)))  float  v8f;

__device__ __forceinline__ u16 f2bf(float f) {
  unsigned u = __float_as_uint(f);
  u += 0x7FFFu + ((u >> 16) & 1u);   // round-to-nearest-even
  return (u16)(u >> 16);
}
__device__ __forceinline__ float bf2f(u16 b) { return __uint_as_float(((unsigned)b) << 16); }
__device__ __forceinline__ float sigm(float x) { return 1.0f / (1.0f + __expf(-x)); }

// A-matrix fragment (16x32 bf16, wave32 layout): lane = 16*hh + m holds row m,
// K = {8*hh .. 8*hh+7} in v[0..7] and {16+8*hh ..} in v[8..15]  -> two b128 LDS loads.
__device__ __forceinline__ v16bf load_a_frag(const u16* row, int k0, int hh) {
  v8bf lo = *(const v8bf*)(row + k0 + 8 * hh);
  v8bf hi = *(const v8bf*)(row + k0 + 16 + 8 * hh);
  return __builtin_shufflevector(lo, hi, 0,1,2,3,4,5,6,7,8,9,10,11,12,13,14,15);
}

__device__ __forceinline__ v16bf load_b_col(const u16* col, int ld, int k0, int hh) {
  v16bf b;
#pragma unroll
  for (int s = 0; s < 16; ++s) {
    const int k = k0 + (s < 8 ? 8 * hh + s : 16 + 8 * hh + (s - 8));
    b[s] = __builtin_bit_cast(__bf16, col[(size_t)k * ld]);
  }
  return b;
}

__device__ __forceinline__ v8f wmma_bf(v16bf a, v16bf b, v8f c) {
  return __builtin_amdgcn_wmma_f32_16x16x32_bf16(false, a, false, b, (short)0, c, false, false);
}

// One mogrify half-step: Z = A(16x512) @ W(512x512) for this wave's 32 columns,
// then T[:, cols] *= 2*sigmoid(Z).   A,T are LDS bf16 arrays (stride LDSS).
__device__ __forceinline__ void mogrify_pass(const u16* __restrict__ A,
                                             const u16* __restrict__ W,
                                             u16* T, int lane, int c0) {
  const int m = lane & 15, hh = lane >> 4;
  const u16* arow = A + m * LDSS;
  v8f z0 = {0,0,0,0,0,0,0,0}, z1 = {0,0,0,0,0,0,0,0};
#pragma unroll 4
  for (int k0 = 0; k0 < 512; k0 += 32) {
    v16bf a = load_a_frag(arow, k0, hh);
    const u16* wk = W + (lane & 15);                  // B-frag: lane holds column lane%16, K in A's order
    z0 = wmma_bf(a, load_b_col(wk + c0, 512, k0, hh),      z0);
    z1 = wmma_bf(a, load_b_col(wk + c0 + 16, 512, k0, hh), z1);
  }
  const int col = c0 + (lane & 15);
#pragma unroll
  for (int r = 0; r < 8; ++r) {
    const int mr = r + 8 * hh;                        // C/D layout: VGPR r -> row r + 8*(lane/16)
    u16* p0 = T + mr * LDSS + col;
    u16* p1 = p0 + 16;
    *p0 = f2bf(2.0f * sigm(z0[r]) * bf2f(*p0));
    *p1 = f2bf(2.0f * sigm(z1[r]) * bf2f(*p1));
  }
}

// Convert all weights f32 -> bf16 into workspace: [Qb | Rb | Wihb | Whhb]
__global__ void wcvt(const float* __restrict__ Q, const float* __restrict__ R,
                     const float* __restrict__ Wih, const float* __restrict__ Whh,
                     u16* __restrict__ ws) {
  const size_t total = 2 * (size_t)NQ + 2 * (size_t)NW;
  for (size_t i = (size_t)blockIdx.x * blockDim.x + threadIdx.x; i < total;
       i += (size_t)gridDim.x * blockDim.x) {
    float v;
    if (i < (size_t)NQ)                 v = Q[i];
    else if (i < 2 * (size_t)NQ)        v = R[i - NQ];
    else if (i < 2 * (size_t)NQ + NW)   v = Wih[i - 2 * (size_t)NQ];
    else                                v = Whh[i - 2 * (size_t)NQ - NW];
    ws[i] = f2bf(v);
  }
}

// Persistent recurrent kernel: 4 blocks (one 16-row batch tile each), 16 waves.
// Wave w owns hidden columns [32w, 32w+32); gate accumulators + Ct live in VGPRs.
__global__ void __launch_bounds__(512)
moglstm_persistent(const float* __restrict__ x,
                   const float* __restrict__ bih,
                   const float* __restrict__ bhh,
                   const u16* __restrict__ wsb,
                   float* __restrict__ out) {
  __shared__ u16 s_x[16 * LDSS];
  __shared__ u16 s_h[16 * LDSS];

  const u16* Qb   = wsb;
  const u16* Rb   = wsb + NQ;
  const u16* Wihb = wsb + 2 * NQ;
  const u16* Whhb = wsb + 2 * NQ + NW;

  const int tid   = threadIdx.x;
  const int lane  = tid & 31;
  const int wv    = tid >> 5;       // 0..15
  const int c0    = wv * 32;        // this wave's column base
  const int m     = lane & 15;
  const int hh    = lane >> 4;
  const int bRow0 = blockIdx.x * 16;

  float* hT  = out + (size_t)Bn * Sn * Hn;
  float* CTo = hT + (size_t)Bn * Hn;

  // h0 = 0 (zero padded LDS rows too)
  for (int r = 0; r < 16; ++r) {
    s_h[r * LDSS + tid] = 0;
    if (tid < (LDSS - 512)) {
      s_h[r * LDSS + 512 + tid] = 0;
      s_x[r * LDSS + 512 + tid] = 0;
    }
  }

  // time-invariant per-lane gate biases (ingate|forget|cell|out column layout)
  float bias[4][2];
#pragma unroll
  for (int g = 0; g < 4; ++g)
#pragma unroll
    for (int j = 0; j < 2; ++j) {
      const int n = g * 512 + c0 + 16 * j + m;
      bias[g][j] = bih[n] + bhh[n];
    }

  v8f ct[2];                        // cell state for this wave's 32 columns
#pragma unroll
  for (int j = 0; j < 2; ++j)
#pragma unroll
    for (int r = 0; r < 8; ++r) ct[j][r] = 0.0f;

  __syncthreads();

  for (int t = 0; t < Sn; ++t) {
    // ---- stage x_t into LDS (bf16), coalesced 512-float rows ----
#pragma unroll 4
    for (int r = 0; r < 16; ++r)
      s_x[r * LDSS + tid] = f2bf(x[((size_t)(bRow0 + r) * Sn + t) * In + tid]);
    if (t + 1 < Sn)   // hint next step's activations toward cache
      __builtin_prefetch(x + ((size_t)(bRow0 + (tid & 15)) * Sn + (t + 1)) * In +
                         ((tid >> 4) << 4), 0, 1);
    __syncthreads();

    // ---- mogrify: x,h,x,h,x (iters 1..5) ----
    mogrify_pass(s_h, Qb, s_x, lane, c0); __syncthreads();
    mogrify_pass(s_x, Rb, s_h, lane, c0); __syncthreads();
    mogrify_pass(s_h, Qb, s_x, lane, c0); __syncthreads();
    mogrify_pass(s_x, Rb, s_h, lane, c0); __syncthreads();
    mogrify_pass(s_h, Qb, s_x, lane, c0); __syncthreads();

    // ---- gates = xt@Wih + ht@Whh + (bih+bhh), 8 tiles/wave, K=512 ----
    v8f acc[4][2];
#pragma unroll
    for (int g = 0; g < 4; ++g)
#pragma unroll
      for (int j = 0; j < 2; ++j)
#pragma unroll
        for (int r = 0; r < 8; ++r) acc[g][j][r] = bias[g][j];

    const u16* axrow = s_x + m * LDSS;
    const u16* ahrow = s_h + m * LDSS;
#pragma unroll 2
    for (int k0 = 0; k0 < 512; k0 += 32) {
      v16bf ax = load_a_frag(axrow, k0, hh);
      v16bf ah = load_a_frag(ahrow, k0, hh);
      const u16* wik = Wihb + (lane & 15);
      const u16* whk = Whhb + (lane & 15);
#pragma unroll
      for (int g = 0; g < 4; ++g)
#pragma unroll
        for (int j = 0; j < 2; ++j) {
          const int n = g * 512 + c0 + 16 * j;
          acc[g][j] = wmma_bf(ax, load_b_col(wik + n, Gn, k0, hh), acc[g][j]);
          acc[g][j] = wmma_bf(ah, load_b_col(whk + n, Gn, k0, hh), acc[g][j]);
        }
    }
    __syncthreads();  // all waves done reading s_x/s_h before we overwrite s_h

    // ---- LSTM cell update (registers) + write h_t ----
#pragma unroll
    for (int j = 0; j < 2; ++j) {
      const int col = c0 + 16 * j + m;
#pragma unroll
      for (int r = 0; r < 8; ++r) {
        const int mr = r + 8 * hh;
        const float it = sigm(acc[0][j][r]);
        const float ft = sigm(acc[1][j][r]);
        const float gt = tanhf(acc[2][j][r]);
        const float ot = sigm(acc[3][j][r]);
        const float c  = ft * ct[j][r] + it * gt;
        ct[j][r] = c;
        const float hv = ot * tanhf(c);
        s_h[mr * LDSS + col] = f2bf(hv);
        out[((size_t)(bRow0 + mr) * Sn + t) * Hn + col] = hv;
        if (t == Sn - 1) {
          hT [(size_t)(bRow0 + mr) * Hn + col] = hv;
          CTo[(size_t)(bRow0 + mr) * Hn + col] = c;
        }
      }
    }
    // next iteration's x-store is ordered by the barrier at top of mogrify chain
  }
}

extern "C" void kernel_launch(void* const* d_in, const int* in_sizes, int n_in,
                              void* d_out, int out_size, void* d_ws, size_t ws_size,
                              hipStream_t stream) {
  (void)in_sizes; (void)n_in; (void)out_size; (void)ws_size;
  const float* x   = (const float*)d_in[0];
  const float* Wih = (const float*)d_in[1];
  const float* Whh = (const float*)d_in[2];
  const float* bih = (const float*)d_in[3];
  const float* bhh = (const float*)d_in[4];
  const float* Q   = (const float*)d_in[5];
  const float* R   = (const float*)d_in[6];
  u16* wsb = (u16*)d_ws;   // needs 2*(NQ+NW)*2 bytes = ~5.25 MB

  wcvt<<<2560, 512, 0, stream>>>(Q, R, Wih, Whh, wsb);
  moglstm_persistent<<<4, 512, 0, stream>>>(x, bih, bhh, wsb, (float*)d_out);
}
